// MultiHeadAttention_8589935261
// MI455X (gfx1250) — compile-verified, matrix operand order checked by reading, not run on hardware
//
#include <hip/hip_runtime.h>
#include <hip/hip_bf16.h>

// Problem constants (match reference)
#define NLOC   4096
#define DMODEL 512
#define NH     4
#define DK     128
#define DV     256
#define KT     64      // keys per attention iteration

typedef _Float16 v16h  __attribute__((ext_vector_type(16)));
typedef _Float16 half8 __attribute__((ext_vector_type(8)));
typedef float    v8f   __attribute__((ext_vector_type(8)));
typedef __attribute__((address_space(3))) _Float16 lds_f16;

union HF16 { v16h v; half8 h[2]; };

__device__ __forceinline__ v8f wmma16(v16h a, v16h b, v8f c) {
    // D = A(16x32 f16) x B(32x16 f16) + C(16x16 f32)
    return __builtin_amdgcn_wmma_f32_16x16x32_f16(
        /*neg_a=*/false, a, /*neg_b=*/false, b,
        /*c_mod=*/(short)0, c, /*reuse_a=*/false, /*reuse_b=*/false);
}

// Load a 16x32 f16 fragment from a row-major matrix p (leading dim ld).
// CDNA5 16-bit operand layout: lane<16 holds K in {0..7, 16..23} of its row,
// lane>=16 holds K in {8..15, 24..31}. Works for global or LDS pointers.
__device__ __forceinline__ v16h load_frag(const _Float16* __restrict__ p,
                                          int ld, int row0, int k0, int lane) {
    const int r    = lane & 15;
    const int koff = (lane < 16) ? 0 : 8;
    const _Float16* base = p + (size_t)(row0 + r) * ld + (size_t)(k0 + koff);
    HF16 f;
    f.h[0] = *(const half8*)(base);       // K = k0+koff    .. +7   (b128)
    f.h[1] = *(const half8*)(base + 16);  // K = k0+koff+16 .. +23  (b128)
    return f.v;
}

// ---- CDNA5 async LDS-DMA (no destination VGPRs, tracked by ASYNCcnt) ------
__device__ __forceinline__ unsigned lds_off(const _Float16* p) {
    return (unsigned)(size_t)(lds_f16*)p;          // addrspacecast -> LDS offset
}
__device__ __forceinline__ void async_load_b128(unsigned dst_lds,
                                                const _Float16* src) {
    asm volatile("global_load_async_to_lds_b128 %0, %1, off"
                 :: "v"(dst_lds), "v"((unsigned long long)(size_t)src)
                 : "memory");
}
__device__ __forceinline__ void wait_async0() {
    asm volatile("s_wait_asynccnt 0" ::: "memory");
}

// ---------------------------------------------------------------------------
// Conversion / layout kernels (run once, negligible)
// ---------------------------------------------------------------------------
__global__ void cvt_copy_f32_f16(const float* __restrict__ src,
                                 _Float16* __restrict__ dst, long n) {
    long i = (long)blockIdx.x * blockDim.x + threadIdx.x;
    if (i < n) dst[i] = (_Float16)src[i];
}

// src [B][R][C] f32 -> dst [B][C][R] f16
__global__ void cvt_transpose_f32_f16(const float* __restrict__ src,
                                      _Float16* __restrict__ dst,
                                      int B, int R, int C) {
    long i = (long)blockIdx.x * blockDim.x + threadIdx.x;
    long total = (long)B * R * C;
    if (i >= total) return;
    int c = (int)(i % C);
    long t = i / C;
    int r = (int)(t % R);
    int b = (int)(t / R);
    dst[((size_t)b * C + c) * R + r] = (_Float16)src[i];
}

// ---------------------------------------------------------------------------
// Generic WMMA GEMM:  Out[n][m] = sum_k A[m][k] * B[n][k]
// 32x32 tile per wave (2x2 register blocking).
// ---------------------------------------------------------------------------
__device__ __forceinline__ void store_tile(_Float16* o, const v8f& acc) {
    half8 st;
#pragma unroll
    for (int r = 0; r < 8; ++r) st[r] = (_Float16)acc[r];
    *(half8*)o = st;                       // one b128 store
}
__device__ __forceinline__ void store_tile(float* o, const v8f& acc) {
    *(v8f*)o = acc;                        // two b128 stores
}

template <typename OutT>
__global__ __launch_bounds__(128)
void gemm_btat(const _Float16* __restrict__ A, const _Float16* __restrict__ B,
               OutT* __restrict__ Out, int Kdim, int Mtotal, int Ntotal,
               long long sA, long long sB, long long sO) {
    const int lane = threadIdx.x & 31;
    const int wave = threadIdx.x >> 5;
    const int mt0 = blockIdx.x * 2;                 // two M tiles
    const int nt0 = (blockIdx.y * 4 + wave) * 2;    // two N tiles (wave-uniform)
    if (nt0 * 16 >= Ntotal) return;
    const int z = blockIdx.z;
    A   += (size_t)z * sA;
    B   += (size_t)z * sB;
    Out += (size_t)z * sO;

    v8f acc00 = {}, acc01 = {}, acc10 = {}, acc11 = {};
    for (int k0 = 0; k0 < Kdim; k0 += 32) {
        v16h a0 = load_frag(A, Kdim, mt0 * 16,       k0, lane);
        v16h a1 = load_frag(A, Kdim, (mt0 + 1) * 16, k0, lane);
        v16h b0 = load_frag(B, Kdim, nt0 * 16,       k0, lane);
        v16h b1 = load_frag(B, Kdim, (nt0 + 1) * 16, k0, lane);
        acc00 = wmma16(a0, b0, acc00);
        acc01 = wmma16(a0, b1, acc01);
        acc10 = wmma16(a1, b0, acc10);
        acc11 = wmma16(a1, b1, acc11);
    }
    const int rlo = lane & 15;
    const int clo = (lane < 16) ? 0 : 8;
#pragma unroll
    for (int j = 0; j < 2; ++j) {
        const int row = (nt0 + j) * 16 + rlo;
        OutT* orow = Out + (size_t)row * Mtotal + mt0 * 16 + clo;
        store_tile(orow,      j ? acc01 : acc00);
        store_tile(orow + 16, j ? acc11 : acc10);
    }
}

// ---------------------------------------------------------------------------
// Flash attention with double-buffered async LDS staging.
// 4 waves/block; each wave owns 16 query rows; the block shares K tiles
// (KT x 128) and V^T tiles (256 x KT) staged by GLOBAL_LOAD_ASYNC_TO_LDS
// (zero VGPR cost) while computing on the other buffer.
// qh/kh: [H][N][DK] f16, vT: [H][DV][N] f16.  X out: [N][H*DV] f16.
// ---------------------------------------------------------------------------
#define L2E 1.4426950408889634f

__device__ __forceinline__ void stage_async(_Float16* dK, _Float16* dV,
                                            const _Float16* kp,
                                            const _Float16* vp,
                                            int kb, int tid) {
    // K tile: 64x128 halves, contiguous in global — 8 x b128 per thread slot
#pragma unroll
    for (int j = 0; j < 8; ++j) {
        const int f = j * 128 + tid;                 // half8 slot
        async_load_b128(lds_off(dK + f * 8), kp + (size_t)kb * DK + f * 8);
    }
    // V^T tile: 256 rows x 64 halves (global ld = NLOC) — 16 x b128 per slot
#pragma unroll
    for (int j = 0; j < 16; ++j) {
        const int f   = j * 128 + tid;
        const int row = f >> 3;
        const int c8  = f & 7;
        async_load_b128(lds_off(dV + f * 8),
                        vp + (size_t)row * NLOC + kb + c8 * 8);
    }
}

__global__ __launch_bounds__(128)
void flash_attn(const _Float16* __restrict__ qh, const _Float16* __restrict__ kh,
                const _Float16* __restrict__ vT, _Float16* __restrict__ X) {
    __shared__ _Float16 sK[2][KT * DK];     // 2 x 16 KB
    __shared__ _Float16 sV[2][DV * KT];     // 2 x 32 KB
    __shared__ _Float16 sP[4 * 16 * KT];    // per-wave P, 8 KB
    const int tid  = threadIdx.x;
    const int lane = tid & 31;
    const int wave = tid >> 5;
    const int h  = blockIdx.y;
    const int q0 = (blockIdx.x * 4 + wave) * 16;

    const _Float16* qp = qh + (size_t)h * NLOC * DK;
    const _Float16* kp = kh + (size_t)h * NLOC * DK;
    const _Float16* vp = vT + (size_t)h * DV * NLOC;

    // Q fragments (16 rows x 128), pre-scaled by 1/sqrt(N) = 1/64
    v16h qf[4];
#pragma unroll
    for (int c = 0; c < 4; ++c) {
        qf[c] = load_frag(qp, DK, q0, c * 32, lane);
#pragma unroll
        for (int e = 0; e < 16; ++e) qf[c][e] *= (_Float16)0.015625f;
    }

    float mrow[8], lrow[8];
    v8f O[16];
#pragma unroll
    for (int r = 0; r < 8; ++r) { mrow[r] = -1.0e30f; lrow[r] = 0.0f; }
#pragma unroll
    for (int t = 0; t < 16; ++t) O[t] = v8f{};

    _Float16* myP = sP + wave * (16 * KT);

    // Pre-stage first tile into buffer 0
    stage_async(sK[0], sV[0], kp, vp, 0, tid);

    int buf = 0;
    for (int kb = 0; kb < NLOC; kb += KT, buf ^= 1) {
        wait_async0();          // own async loads into sK/sV[buf] complete
        __syncthreads();        // all waves' loads visible; prev reads done
        if (kb + KT < NLOC)     // overlap: stage next tile while computing
            stage_async(sK[buf ^ 1], sV[buf ^ 1], kp, vp, kb + KT, tid);

        const _Float16* kcur = sK[buf];
        const _Float16* vcur = sV[buf];

        // ---- S = q . k^T for 64 keys (four 16x16 tiles) ------------------
        v8f s0 = {}, s1 = {}, s2 = {}, s3 = {};
#pragma unroll
        for (int c = 0; c < 4; ++c) {
            v16h b0 = load_frag(kcur, DK,  0, c * 32, lane);
            v16h b1 = load_frag(kcur, DK, 16, c * 32, lane);
            v16h b2 = load_frag(kcur, DK, 32, c * 32, lane);
            v16h b3 = load_frag(kcur, DK, 48, c * 32, lane);
            s0 = wmma16(qf[c], b0, s0);
            s1 = wmma16(qf[c], b1, s1);
            s2 = wmma16(qf[c], b2, s2);
            s3 = wmma16(qf[c], b3, s3);
        }
        // ---- online softmax ---------------------------------------------
#pragma unroll
        for (int r = 0; r < 8; ++r) {
            float smax = fmaxf(fmaxf(s0[r], s1[r]), fmaxf(s2[r], s3[r]));
#pragma unroll
            for (int m = 8; m >= 1; m >>= 1)
                smax = fmaxf(smax, __shfl_xor(smax, m, 32));
            const float mnew  = fmaxf(mrow[r], smax);
            const float alpha = exp2f((mrow[r] - mnew) * L2E);
            const float p0 = exp2f((s0[r] - mnew) * L2E);
            const float p1 = exp2f((s1[r] - mnew) * L2E);
            const float p2 = exp2f((s2[r] - mnew) * L2E);
            const float p3 = exp2f((s3[r] - mnew) * L2E);
            float rs = (p0 + p1) + (p2 + p3);
#pragma unroll
            for (int m = 8; m >= 1; m >>= 1) rs += __shfl_xor(rs, m, 32);
            lrow[r] = lrow[r] * alpha + rs;
            mrow[r] = mnew;
#pragma unroll
            for (int t = 0; t < 16; ++t) O[t][r] *= alpha;
            const int prow = r + ((lane < 16) ? 0 : 8);
            _Float16* pr = myP + prow * KT + (lane & 15);
            pr[0]  = (_Float16)p0;
            pr[16] = (_Float16)p1;
            pr[32] = (_Float16)p2;
            pr[48] = (_Float16)p3;
        }
        asm volatile("s_wait_dscnt 0" ::: "memory");  // P visible wave-wide
        // ---- read P back as two 16x32 A fragments ------------------------
        v16h pf0 = load_frag(myP, KT, 0,  0, lane);
        v16h pf1 = load_frag(myP, KT, 0, 32, lane);
        // ---- O += P x V: batch 4 B-frag loads per 4 WMMAs ----------------
#pragma unroll
        for (int t = 0; t < 16; t += 2) {
            v16h v00 = load_frag(vcur, KT, t * 16,        0, lane);
            v16h v01 = load_frag(vcur, KT, t * 16,       32, lane);
            v16h v10 = load_frag(vcur, KT, (t + 1) * 16,  0, lane);
            v16h v11 = load_frag(vcur, KT, (t + 1) * 16, 32, lane);
            O[t]     = wmma16(pf0, v00, O[t]);
            O[t]     = wmma16(pf1, v01, O[t]);
            O[t + 1] = wmma16(pf0, v10, O[t + 1]);
            O[t + 1] = wmma16(pf1, v11, O[t + 1]);
        }
    }

    // ---- epilogue: normalize, scatter into X[n][h*DV + vf] ---------------
#pragma unroll
    for (int r = 0; r < 8; ++r) {
        const float inv = 1.0f / lrow[r];
        const int row = r + ((lane < 16) ? 0 : 8);
        _Float16* xr = X + (size_t)(q0 + row) * (NH * DV) + h * DV + (lane & 15);
#pragma unroll
        for (int t = 0; t < 16; ++t)
            xr[t * 16] = (_Float16)(O[t][r] * inv);
    }
}

// ---------------------------------------------------------------------------
// Host-side orchestration
// ---------------------------------------------------------------------------
extern "C" void kernel_launch(void* const* d_in, const int* in_sizes, int n_in,
                              void* d_out, int out_size, void* d_ws, size_t ws_size,
                              hipStream_t stream) {
    const float* Q  = (const float*)d_in[0];
    const float* K  = (const float*)d_in[1];
    const float* V  = (const float*)d_in[2];
    const float* Wq = (const float*)d_in[3];
    const float* Wk = (const float*)d_in[4];
    const float* Wv = (const float*)d_in[5];
    const float* Wo = (const float*)d_in[6];
    float* out = (float*)d_out;

    // Workspace layout (f16 elements)
    _Float16* ws = (_Float16*)d_ws;
    size_t off = 0;
    _Float16* Qh  = ws + off; off += (size_t)NLOC * DMODEL;        // [N][512]
    _Float16* Kh  = ws + off; off += (size_t)NLOC * DMODEL;
    _Float16* Vh  = ws + off; off += (size_t)NLOC * DMODEL;
    _Float16* WqT = ws + off; off += (size_t)NH * DK * DMODEL;     // [H][128][512]
    _Float16* WkT = ws + off; off += (size_t)NH * DK * DMODEL;
    _Float16* WvT = ws + off; off += (size_t)NH * DV * DMODEL;     // [H][256][512]
    _Float16* WoH = ws + off; off += (size_t)DMODEL * NH * DV;     // [512][1024]
    _Float16* qhv = ws + off; off += (size_t)NH * NLOC * DK;       // [H][N][128]
    _Float16* khv = ws + off; off += (size_t)NH * NLOC * DK;
    _Float16* vTv = ws + off; off += (size_t)NH * DV * NLOC;       // [H][256][N]
    _Float16* Xv  = ws + off; off += (size_t)NLOC * NH * DV;       // [N][1024]

    const int T = 256;
    const long nAct = (long)NLOC * DMODEL;
    cvt_copy_f32_f16<<<(nAct + T - 1) / T, T, 0, stream>>>(Q, Qh, nAct);
    cvt_copy_f32_f16<<<(nAct + T - 1) / T, T, 0, stream>>>(K, Kh, nAct);
    cvt_copy_f32_f16<<<(nAct + T - 1) / T, T, 0, stream>>>(V, Vh, nAct);
    const long nWo = (long)DMODEL * NH * DV;
    cvt_copy_f32_f16<<<(nWo + T - 1) / T, T, 0, stream>>>(Wo, WoH, nWo);

    cvt_transpose_f32_f16<<<((long)NH * DMODEL * DK + T - 1) / T, T, 0, stream>>>(
        Wq, WqT, NH, DMODEL, DK);
    cvt_transpose_f32_f16<<<((long)NH * DMODEL * DK + T - 1) / T, T, 0, stream>>>(
        Wk, WkT, NH, DMODEL, DK);
    cvt_transpose_f32_f16<<<((long)NH * DMODEL * DV + T - 1) / T, T, 0, stream>>>(
        Wv, WvT, NH, DMODEL, DV);

    // q/k projections: qh[h][n][dk] = sum_d WqT[h][dk][d] * Qh[n][d]
    {
        dim3 grid(DK / 32, NLOC / 128, NH);      // (4, 32, 4)
        gemm_btat<_Float16><<<grid, 128, 0, stream>>>(
            WqT, Qh, qhv, DMODEL, DK, NLOC,
            (long long)DK * DMODEL, 0LL, (long long)NLOC * DK);
        gemm_btat<_Float16><<<grid, 128, 0, stream>>>(
            WkT, Kh, khv, DMODEL, DK, NLOC,
            (long long)DK * DMODEL, 0LL, (long long)NLOC * DK);
    }
    // v projection, transposed output: vT[h][vf][n] = sum_d Vh[n][d] * WvT[h][vf][d]
    {
        dim3 grid(NLOC / 32, DV / 128, NH);      // (128, 2, 4)
        gemm_btat<_Float16><<<grid, 128, 0, stream>>>(
            Vh, WvT, vTv, DMODEL, NLOC, DV,
            0LL, (long long)DV * DMODEL, (long long)DV * NLOC);
    }
    // fused attention: X[n][h*DV+vf]
    {
        dim3 grid(NLOC / 64, NH);                // (64, 4), 4 waves/block
        flash_attn<<<grid, 128, 0, stream>>>(qhv, khv, vTv, Xv);
    }
    // output projection: out[n][m] = sum_j X[n][j] * WoH[m][j]
    {
        dim3 grid(DMODEL / 32, NLOC / 128, 1);   // (16, 32, 1)
        gemm_btat<float><<<grid, 128, 0, stream>>>(
            WoH, Xv, out, NH * DV, DMODEL, NLOC, 0LL, 0LL, 0LL);
    }
}
